// AxialShift_44324062494783
// MI455X (gfx1250) — matrix operand order repaired and checked by reading, not compiled, not run on hardware
//
#include <hip/hip_runtime.h>

// ---------------------------------------------------------------------------
// Problem constants
// ---------------------------------------------------------------------------
#define C_DIM   160                    // channels (M of weights, K of GEMMs)
#define NSP     110592                 // 48*48*48 spatial positions
#define AX48    48
#define KBLK    5                      // 160 / 32  (WMMA K blocks)
#define NTIL    10                     // 160 / 16  (WMMA N tiles = out channels)
#define NFRAG   (KBLK * NTIL)          // 50 B-fragments per weight matrix
#define WGROWS  128                    // 8 waves * 16 spatial rows per workgroup
#define NWAVES  8
#define NELEM   17694720               // 160 * 110592
#define G_BLOCKS (NSP / WGROWS)        // 864

typedef __attribute__((ext_vector_type(16))) __bf16 v16bf;
typedef __attribute__((ext_vector_type(8)))  __bf16 v8bf;
typedef __attribute__((ext_vector_type(8)))  float  v8f;

static __device__ __forceinline__ float gelu_erf(float x) {
    return 0.5f * x * (1.0f + erff(x * 0.70710678118654752440f));
}

static __device__ __forceinline__ v16bf cat8(v8bf a0, v8bf a1) {
    return __builtin_shufflevector(a0, a1, 0,1,2,3,4,5,6,7,8,9,10,11,12,13,14,15);
}

static __device__ __forceinline__ v16bf zero16bf() {
    v16bf z;
#pragma unroll
    for (int i = 0; i < 16; ++i) z[i] = (__bf16)0.0f;
    return z;
}

// A-fragment (16x32 bf16): lane m(0..15) holds row m, K {0..7,16..23};
// lane 16+m holds row m, K {8..15,24..31}. Two contiguous 16B global loads.
static __device__ __forceinline__ v16bf load_a_frag(const __bf16* __restrict__ rowptr,
                                                    int kb, int chunkoff) {
    const __bf16* p = rowptr + kb * 32 + chunkoff;
    v8bf a0 = *(const v8bf*)(p);
    v8bf a1 = *(const v8bf*)(p + 16);
    return cat8(a0, a1);
}

// B-fragment from LDS. Layout per fragment (512 elements = 1KB):
//   chunk0 pool: lane L's first  8 elems at  fbase + L*8
//   chunk1 pool: lane L's second 8 elems at  fbase + 256 + L*8
// => each ds_load_b128 has lanes at stride 16B: minimal 2-way bank pattern.
static __device__ __forceinline__ v16bf load_b_frag(const __bf16* wlds, int frag, int lane) {
    const __bf16* fb = wlds + frag * 512;
    v8bf b0 = *(const v8bf*)(fb + lane * 8);
    v8bf b1 = *(const v8bf*)(fb + 256 + lane * 8);
    return cat8(b0, b1);
}

// ---------------------------------------------------------------------------
// K0a: swizzle+convert the 4 pre-known weights (w1,w21,w22,w23) into the
// B-fragment layout above:
//   e = frag*512 + q*256 + lane*8 + j   (i = q*8 + j)
//   value = W[o = nt*16 + lane%16][c = kb*32 + q*16 + (lane/16)*8 + j]
// ---------------------------------------------------------------------------
__global__ void kPrepW(const float* __restrict__ w1,  const float* __restrict__ w21,
                       const float* __restrict__ w22, const float* __restrict__ w23,
                       __bf16* __restrict__ wswz) {
    int e = blockIdx.x * blockDim.x + threadIdx.x;
    if (e >= 4 * C_DIM * C_DIM) return;
    int wsel = e / (C_DIM * C_DIM);
    int r    = e % (C_DIM * C_DIM);
    const float* src = (wsel == 0) ? w1 : (wsel == 1) ? w21 : (wsel == 2) ? w22 : w23;
    int j    = r & 7;
    int lane = (r >> 3) & 31;
    int q    = (r >> 8) & 1;
    int frag = r >> 9;
    int nt   = frag % NTIL;
    int kb   = frag / NTIL;
    int i    = q * 8 + j;
    int o    = nt * 16 + (lane & 15);
    int c    = kb * 32 + q * 16 + ((lane >> 4) << 3) + j;
    wswz[e] = (__bf16)src[o * C_DIM + c];
}

// ---------------------------------------------------------------------------
// K0b: transpose x from [c][p] fp32 to channel-contiguous [p][c] bf16
// ---------------------------------------------------------------------------
__global__ __launch_bounds__(256)
void kTransX(const float* __restrict__ x, __bf16* __restrict__ xb) {
    __shared__ float t[32][65];
    const int cb = blockIdx.y * 32;
    const int pb = blockIdx.x * 64;
    for (int i = threadIdx.x; i < 32 * 64; i += 256) {
        int pl = i & 63, cl = i >> 6;
        t[cl][pl] = x[(size_t)(cb + cl) * NSP + pb + pl];
    }
    __syncthreads();
    for (int i = threadIdx.x; i < 32 * 64; i += 256) {
        int cl = i & 31, pl = i >> 5;
        xb[(size_t)(pb + pl) * C_DIM + cb + cl] = (__bf16)t[cl][pl];
    }
}

// ---------------------------------------------------------------------------
// GEMM1: y1[p][o] = sum_c xb[p][c] * w1[o][c] + b1[o]  (bf16 WMMA, f32 acc)
// B-fragments are double-buffered so ds_load latency hides under the WMMAs.
// Epilogue: bf16 store + GroupNorm-1 partial stats (sum, sumsq) via atomics.
// ---------------------------------------------------------------------------
__global__ __launch_bounds__(256)
void kGemm1(const __bf16* __restrict__ act, const __bf16* __restrict__ wswz,
            const float* __restrict__ bias, __bf16* __restrict__ out,
            float* __restrict__ stats) {
    __shared__ __align__(32) __bf16 wlds[C_DIM * C_DIM];
    __shared__ float red[NWAVES * 2];
    {
        const uint32_t* s = (const uint32_t*)wswz;
        uint32_t* d = (uint32_t*)wlds;
        for (int i = threadIdx.x; i < C_DIM * C_DIM / 2; i += 256) d[i] = s[i];
    }
    __syncthreads();

    const int lane = threadIdx.x & 31;
    const int wave = threadIdx.x >> 5;
    const int half = lane >> 4;
    const int ln16 = lane & 15;
    const int p0   = blockIdx.x * WGROWS;
    const int arow = p0 + wave * 16 + ln16;
    const __bf16* aptr = act + (size_t)arow * C_DIM;
    const int chunkoff = half * 8;

    v8f acc[NTIL] = {};
    v16bf b = load_b_frag(wlds, 0, lane);
#pragma unroll
    for (int kb = 0; kb < KBLK; ++kb) {
        v16bf a = load_a_frag(aptr, kb, chunkoff);
#pragma unroll
        for (int nt = 0; nt < NTIL; ++nt) {
            const int nxt = kb * NTIL + nt + 1;
            v16bf bn = (nxt < NFRAG) ? load_b_frag(wlds, nxt, lane) : b;
            acc[nt] = __builtin_amdgcn_wmma_f32_16x16x32_bf16(
                false, a, false, b, (short)0, acc[nt], false, false);
            b = bn;
        }
    }

    float lsum = 0.f, lsq = 0.f;
#pragma unroll
    for (int nt = 0; nt < NTIL; ++nt) {
        const int c = nt * 16 + ln16;
        const float bo = bias[c];
#pragma unroll
        for (int r = 0; r < 8; ++r) {
            float v = acc[nt][r] + bo;
            lsum += v; lsq += v * v;
            int p = p0 + wave * 16 + half * 8 + r;
            out[(size_t)p * C_DIM + c] = (__bf16)v;
        }
    }
    for (int off = 16; off > 0; off >>= 1) {
        lsum += __shfl_xor(lsum, off, 32);
        lsq  += __shfl_xor(lsq,  off, 32);
    }
    if (lane == 0) { red[wave * 2] = lsum; red[wave * 2 + 1] = lsq; }
    __syncthreads();
    if (threadIdx.x == 0) {
        float s = 0.f, q = 0.f;
        for (int i = 0; i < NWAVES; ++i) { s += red[i * 2]; q += red[i * 2 + 1]; }
        atomicAdd(&stats[0], s);
        atomicAdd(&stats[1], q);
    }
}

// ---------------------------------------------------------------------------
// Elementwise: h = gelu(groupnorm1(y1))   (bf16 -> bf16), stats from ws
// ---------------------------------------------------------------------------
__global__ __launch_bounds__(256)
void kAct1(const __bf16* __restrict__ y1, const float* __restrict__ g1,
           const float* __restrict__ be1, const float* __restrict__ stats,
           __bf16* __restrict__ h) {
    const float inv  = 1.0f / (float)NELEM;
    const float mu   = stats[0] * inv;
    const float var  = stats[1] * inv - mu * mu;
    const float rstd = rsqrtf(var + 1e-5f);
    size_t idx = ((size_t)blockIdx.x * blockDim.x + threadIdx.x) * 4;
    if (idx >= (size_t)NELEM) return;
    int c = (int)(idx % C_DIM);
#pragma unroll
    for (int j = 0; j < 4; ++j) {
        float v = (float)y1[idx + j];
        float z = (v - mu) * rstd * g1[c + j] + be1[c + j];
        h[idx + j] = (__bf16)gelu_erf(z);
    }
}

// ---------------------------------------------------------------------------
// GEMM2 (fused): y2 = gelu(W21*shiftD(h)+b21) + gelu(W22*shiftH(h)+b22)
//                   + gelu(W23*shiftW(h)+b23)
// The shift's 32-channel chunks == WMMA K-blocks, so each K-step reads the
// A row at p' = p - s*stride (s = kb-2), zero outside [0,48) on that axis.
// Epilogue: bf16 store + GroupNorm-2 partial stats.
// ---------------------------------------------------------------------------
__global__ __launch_bounds__(256)
void kGemm2(const __bf16* __restrict__ act, const __bf16* __restrict__ wswz3,
            const float* __restrict__ b21, const float* __restrict__ b22,
            const float* __restrict__ b23, __bf16* __restrict__ out,
            float* __restrict__ stats2) {
    __shared__ __align__(32) __bf16 wlds[C_DIM * C_DIM];
    __shared__ float red[NWAVES * 2];

    const int lane = threadIdx.x & 31;
    const int wave = threadIdx.x >> 5;
    const int half = lane >> 4;
    const int ln16 = lane & 15;
    const int p0   = blockIdx.x * WGROWS;
    const int arow = p0 + wave * 16 + ln16;
    const int chunkoff = half * 8;

    const int dw = arow % AX48;
    const int hh = (arow / AX48) % AX48;
    const int dd = arow / (AX48 * AX48);

    float sum[NTIL][8] = {};

    for (int ax = 0; ax < 3; ++ax) {
        __syncthreads();
        {   // load this axis' weight matrix into LDS (B-fragment order)
            const uint32_t* s = (const uint32_t*)(wswz3 + (size_t)ax * C_DIM * C_DIM);
            uint32_t* d = (uint32_t*)wlds;
            for (int i = threadIdx.x; i < C_DIM * C_DIM / 2; i += 256) d[i] = s[i];
        }
        __syncthreads();

        const int stride = (ax == 0) ? AX48 * AX48 : (ax == 1) ? AX48 : 1;
        const int coord  = (ax == 0) ? dd : (ax == 1) ? hh : dw;
        const float* bb  = (ax == 0) ? b21 : (ax == 1) ? b22 : b23;

        v8f acc[NTIL] = {};
        v16bf b = load_b_frag(wlds, 0, lane);
#pragma unroll
        for (int kb = 0; kb < KBLK; ++kb) {
            const int s = kb - 2;                 // channel-chunk shift
            const int c2 = coord - s;
            v16bf a;
            if (c2 >= 0 && c2 < AX48) {
                const __bf16* rowptr = act + (size_t)(arow - s * stride) * C_DIM;
                a = load_a_frag(rowptr, kb, chunkoff);
            } else {
                a = zero16bf();
            }
#pragma unroll
            for (int nt = 0; nt < NTIL; ++nt) {
                const int nxt = kb * NTIL + nt + 1;
                v16bf bn = (nxt < NFRAG) ? load_b_frag(wlds, nxt, lane) : b;
                acc[nt] = __builtin_amdgcn_wmma_f32_16x16x32_bf16(
                    false, a, false, b, (short)0, acc[nt], false, false);
                b = bn;
            }
        }
#pragma unroll
        for (int nt = 0; nt < NTIL; ++nt) {
            const float bo = bb[nt * 16 + ln16];
#pragma unroll
            for (int r = 0; r < 8; ++r)
                sum[nt][r] += gelu_erf(acc[nt][r] + bo);
        }
    }

    float lsum = 0.f, lsq = 0.f;
#pragma unroll
    for (int nt = 0; nt < NTIL; ++nt) {
        const int c = nt * 16 + ln16;
#pragma unroll
        for (int r = 0; r < 8; ++r) {
            float v = sum[nt][r];
            lsum += v; lsq += v * v;
            int p = p0 + wave * 16 + half * 8 + r;
            out[(size_t)p * C_DIM + c] = (__bf16)v;
        }
    }
    for (int off = 16; off > 0; off >>= 1) {
        lsum += __shfl_xor(lsum, off, 32);
        lsq  += __shfl_xor(lsq,  off, 32);
    }
    if (lane == 0) { red[wave * 2] = lsum; red[wave * 2 + 1] = lsq; }
    __syncthreads();
    if (threadIdx.x == 0) {
        float s = 0.f, q = 0.f;
        for (int i = 0; i < NWAVES; ++i) { s += red[i * 2]; q += red[i * 2 + 1]; }
        atomicAdd(&stats2[0], s);
        atomicAdd(&stats2[1], q);
    }
}

// ---------------------------------------------------------------------------
// Fold GroupNorm-2 into W3/b3 (GN affine is linear):
//   w3'[o][c] = w3[o][c]*g2[c]*rstd        (bf16, B-fragment swizzled)
//   b3'[o]    = b3[o] + sum_c w3[o][c]*(be2[c] - mu*rstd*g2[c])
// ---------------------------------------------------------------------------
__global__ void kFoldBias(const float* __restrict__ w3, const float* __restrict__ g2,
                          const float* __restrict__ be2, const float* __restrict__ b3,
                          const float* __restrict__ stats2, float* __restrict__ b3p) {
    int o = threadIdx.x;
    if (o >= C_DIM) return;
    const float inv  = 1.0f / (float)NELEM;
    const float mu   = stats2[0] * inv;
    const float var  = stats2[1] * inv - mu * mu;
    const float rstd = rsqrtf(var + 1e-5f);
    float acc = b3[o];
    for (int c = 0; c < C_DIM; ++c)
        acc += w3[o * C_DIM + c] * (be2[c] - mu * rstd * g2[c]);
    b3p[o] = acc;
}

__global__ void kFoldW(const float* __restrict__ w3, const float* __restrict__ g2,
                       const float* __restrict__ stats2, __bf16* __restrict__ w3swz) {
    int r = blockIdx.x * blockDim.x + threadIdx.x;
    if (r >= C_DIM * C_DIM) return;
    const float inv  = 1.0f / (float)NELEM;
    const float mu   = stats2[0] * inv;
    const float var  = stats2[1] * inv - mu * mu;
    const float rstd = rsqrtf(var + 1e-5f);
    int j    = r & 7;
    int lane = (r >> 3) & 31;
    int q    = (r >> 8) & 1;
    int frag = r >> 9;
    int nt   = frag % NTIL;
    int kb   = frag / NTIL;
    int i    = q * 8 + j;
    int o    = nt * 16 + (lane & 15);
    int c    = kb * 32 + q * 16 + ((lane >> 4) << 3) + j;
    w3swz[r] = (__bf16)(w3[o * C_DIM + c] * g2[c] * rstd);
}

// ---------------------------------------------------------------------------
// GEMM3: out[o][p] = sum_c y2[p][c] * w3'[o][c] + b3'[o], fp32 output in the
// reference (B,C,D,H,W) layout via LDS transpose for coalesced stores.
// LDS buffer is reused: weights (50KB) during compute, tile (80KB) after.
// ---------------------------------------------------------------------------
__global__ __launch_bounds__(256)
void kGemm3(const __bf16* __restrict__ act, const __bf16* __restrict__ wswz,
            const float* __restrict__ bias, float* __restrict__ outp) {
    __shared__ __align__(32) unsigned char smem[WGROWS * C_DIM * sizeof(float)];
    __bf16* wlds = (__bf16*)smem;
    float*  olds = (float*)smem;
    {
        const uint32_t* s = (const uint32_t*)wswz;
        uint32_t* d = (uint32_t*)wlds;
        for (int i = threadIdx.x; i < C_DIM * C_DIM / 2; i += 256) d[i] = s[i];
    }
    __syncthreads();

    const int lane = threadIdx.x & 31;
    const int wave = threadIdx.x >> 5;
    const int half = lane >> 4;
    const int ln16 = lane & 15;
    const int p0   = blockIdx.x * WGROWS;
    const int arow = p0 + wave * 16 + ln16;
    const __bf16* aptr = act + (size_t)arow * C_DIM;
    const int chunkoff = half * 8;

    v8f acc[NTIL] = {};
    v16bf b = load_b_frag(wlds, 0, lane);
#pragma unroll
    for (int kb = 0; kb < KBLK; ++kb) {
        v16bf a = load_a_frag(aptr, kb, chunkoff);
#pragma unroll
        for (int nt = 0; nt < NTIL; ++nt) {
            const int nxt = kb * NTIL + nt + 1;
            v16bf bn = (nxt < NFRAG) ? load_b_frag(wlds, nxt, lane) : b;
            acc[nt] = __builtin_amdgcn_wmma_f32_16x16x32_bf16(
                false, a, false, b, (short)0, acc[nt], false, false);
            b = bn;
        }
    }
    __syncthreads();   // all waves done reading weights before LDS reuse

#pragma unroll
    for (int nt = 0; nt < NTIL; ++nt) {
        const int c = nt * 16 + ln16;
        const float bo = bias[c];
#pragma unroll
        for (int r = 0; r < 8; ++r) {
            int pl = wave * 16 + half * 8 + r;
            olds[pl * C_DIM + c] = acc[nt][r] + bo;
        }
    }
    __syncthreads();

    for (int i = threadIdx.x; i < WGROWS * C_DIM; i += 256) {
        int pl = i % WGROWS;
        int c  = i / WGROWS;
        outp[(size_t)c * NSP + p0 + pl] = olds[pl * C_DIM + c];
    }
}

// ---------------------------------------------------------------------------
// Launcher
// ---------------------------------------------------------------------------
extern "C" void kernel_launch(void* const* d_in, const int* in_sizes, int n_in,
                              void* d_out, int out_size, void* d_ws, size_t ws_size,
                              hipStream_t stream) {
    const float* x   = (const float*)d_in[0];
    const float* w1  = (const float*)d_in[1];
    const float* b1  = (const float*)d_in[2];
    const float* g1  = (const float*)d_in[3];
    const float* be1 = (const float*)d_in[4];
    const float* w21 = (const float*)d_in[5];
    const float* b21 = (const float*)d_in[6];
    const float* w22 = (const float*)d_in[7];
    const float* b22 = (const float*)d_in[8];
    const float* w23 = (const float*)d_in[9];
    const float* b23 = (const float*)d_in[10];
    const float* g2  = (const float*)d_in[11];
    const float* be2 = (const float*)d_in[12];
    const float* w3  = (const float*)d_in[13];
    const float* b3  = (const float*)d_in[14];
    float* out = (float*)d_out;

    char* wsB = (char*)d_ws;
    size_t off = 0;
    auto take = [&](size_t bytes) {
        size_t o = off;
        off += (bytes + 255) & ~(size_t)255;
        return o;
    };
    float*  stats  = (float*)(wsB + take(4 * sizeof(float)));           // [sum1,sq1,sum2,sq2]
    __bf16* wswz   = (__bf16*)(wsB + take(4ull * C_DIM * C_DIM * 2));   // w1,w21,w22,w23 swizzled
    __bf16* w3swz  = (__bf16*)(wsB + take((size_t)C_DIM * C_DIM * 2));
    float*  b3p    = (float*)(wsB + take(C_DIM * sizeof(float)));
    __bf16* xb     = (__bf16*)(wsB + take((size_t)NELEM * 2));          // x bf16 [p][c]; reused as y2
    __bf16* y1     = (__bf16*)(wsB + take((size_t)NELEM * 2));
    __bf16* hb     = (__bf16*)(wsB + take((size_t)NELEM * 2));
    __bf16* y2     = xb;  // x is dead after GEMM1

    hipMemsetAsync(stats, 0, 4 * sizeof(float), stream);

    kPrepW<<<(4 * C_DIM * C_DIM + 255) / 256, 256, 0, stream>>>(w1, w21, w22, w23, wswz);
    kTransX<<<dim3(NSP / 64, C_DIM / 32), 256, 0, stream>>>(x, xb);
    kGemm1<<<G_BLOCKS, 256, 0, stream>>>(xb, wswz, b1, y1, stats);
    kAct1<<<(NELEM / 4 + 255) / 256, 256, 0, stream>>>(y1, g1, be1, stats, hb);
    kGemm2<<<G_BLOCKS, 256, 0, stream>>>(hb, wswz + (size_t)C_DIM * C_DIM,
                                         b21, b22, b23, y2, stats + 2);
    kFoldBias<<<1, C_DIM, 0, stream>>>(w3, g2, be2, b3, stats + 2, b3p);
    kFoldW<<<(C_DIM * C_DIM + 255) / 256, 256, 0, stream>>>(w3, g2, stats + 2, w3swz);
    kGemm3<<<G_BLOCKS, 256, 0, stream>>>(y2, w3swz, b3p, out);
}
